// AtomTransformer_50010599195182
// MI455X (gfx1250) — compile-verified, matrix operand order checked by reading, not run on hardware
//
#include <hip/hip_runtime.h>
#include <hip/hip_bf16.h>

// ---------------------------------------------------------------------------
// AtomTransformer on gfx1250 (MI455X): token-level f16 WMMA GEMMs + windowed
// attention (4 waves per (batch, window) block, one head per wave) with the
// pair-bias projection fused into attention via a shared LN(pair_cond) f16
// precompute (layer-invariant -> computed once).
// ---------------------------------------------------------------------------

#define B_     4
#define N_     8100
#define D_     128
#define H_     4
#define DH_    32
#define WQ_    64
#define WK_    128
#define NWIN_  127          // ceil(8100/64)
#define NTOK_  8128         // NWIN_*WQ_
#define MT_    (B_*NTOK_)   // 32512 token rows
#define FF_    256
#define DEPTH_ 3
#define DP_    16
#define LDP_   144          // padded LDS row stride in halves (288B = 9*32B)

typedef _Float16 h16v __attribute__((ext_vector_type(16)));
typedef float    f8v  __attribute__((ext_vector_type(8)));

__device__ __forceinline__ f8v wmma16(h16v a, h16v b, f8v c) {
  // D = A(16x32 f16) * B(32x16 f16) + C(16x16 f32)
  return __builtin_amdgcn_wmma_f32_16x16x32_f16(
      /*neg_a=*/false, a, /*neg_b=*/false, b,
      /*c_mod=*/(short)0, c, /*reuse_a=*/false, /*reuse_b=*/false);
}

// Epilogue modes
enum { EP_F32 = 0, EP_F16, EP_BIAS, EP_SIG, EP_SIGBIAS, EP_SILU, EP_MUL, EP_MULADD };

// ---------------------------------------------------------------------------
// Generic WMMA GEMM:  C(M x 128) = epilogue( A(M x K) @ W(K x 128) )
// WT is the weight pre-transposed to (128 x K) f16, row-major (N-major).
// Block: 256 threads = 8 waves, each wave owns 16 rows; block tile 128x128.
// Weight chunk staged in LDS with a 144-half padded stride so the B-fragment
// reads (16 lanes hitting 16 consecutive rows) rotate across banks.
// ---------------------------------------------------------------------------
template<int K, int EPI>
__global__ __launch_bounds__(256) void gemm_wmma_k(
    const _Float16* __restrict__ A,
    const _Float16* __restrict__ WT,     // 128 x K (N-major)
    const float*    __restrict__ bias,   // 128 (or nullptr)
    const _Float16* __restrict__ E1,     // extra elementwise operand (or nullptr)
    const _Float16* __restrict__ E2,     // extra elementwise operand (or nullptr)
    void* __restrict__ Cout, int ldc)
{
  __shared__ _Float16 lW[128 * LDP_];    // 36KB padded weight stage

  const int tid  = threadIdx.x;
  const int wv   = tid >> 5;
  const int lane = tid & 31;
  const int l16  = lane & 15;
  const int hi   = lane >> 4;
  const int rowBase = blockIdx.x * 128 + wv * 16;

  f8v acc[8];
  #pragma unroll
  for (int i = 0; i < 8; i++) acc[i] = (f8v){0.f,0.f,0.f,0.f,0.f,0.f,0.f,0.f};

  for (int kc = 0; kc < K; kc += 128) {
    // cooperative load of the weight chunk: lW[n*LDP + k] = WT[n*K + kc + k]
    for (int i = tid; i < 128 * 16; i += 256) {         // 8 halves per item
      int n = i >> 4, k = (i & 15) * 8;
      *(uint4*)&lW[n * LDP_ + k] = *(const uint4*)&WT[(size_t)n * K + kc + k];
    }
    __syncthreads();

    #pragma unroll
    for (int ks = 0; ks < 4; ks++) {
      // A fragment: lane (l16,hi) -> row l16, K-halves [ks*32 + hi*16, +16)
      h16v af = *(const h16v*)&A[(size_t)(rowBase + l16) * K + kc + ks * 32 + hi * 16];
      #pragma unroll
      for (int nt = 0; nt < 8; nt++) {
        h16v bf = *(const h16v*)&lW[(nt * 16 + l16) * LDP_ + ks * 32 + hi * 16];
        acc[nt] = wmma16(af, bf, acc[nt]);
      }
    }
    __syncthreads();
  }

  // Epilogue.  C layout: vgpr r, lane -> row = r + 8*hi, col = l16 (per 16x16 tile)
  #pragma unroll
  for (int nt = 0; nt < 8; nt++) {
    int col = nt * 16 + l16;
    #pragma unroll
    for (int r = 0; r < 8; r++) {
      int row = rowBase + r + hi * 8;
      float v = acc[nt][r];
      size_t o = (size_t)row * ldc + col;
      if constexpr (EPI == EP_F32) {
        ((float*)Cout)[o] = v;
      } else if constexpr (EPI == EP_F16) {
        ((_Float16*)Cout)[o] = (_Float16)v;
      } else if constexpr (EPI == EP_BIAS) {
        ((_Float16*)Cout)[o] = (_Float16)(v + bias[col]);
      } else if constexpr (EPI == EP_SIG) {
        ((_Float16*)Cout)[o] = (_Float16)(1.f / (1.f + __expf(-v)));
      } else if constexpr (EPI == EP_SIGBIAS) {
        ((_Float16*)Cout)[o] = (_Float16)(1.f / (1.f + __expf(-(v + bias[col]))));
      } else if constexpr (EPI == EP_SILU) {
        ((_Float16*)Cout)[o] = (_Float16)(v / (1.f + __expf(-v)));
      } else if constexpr (EPI == EP_MUL) {
        ((_Float16*)Cout)[o] = (_Float16)((float)E1[o] * v);
      } else if constexpr (EPI == EP_MULADD) {
        ((_Float16*)Cout)[o] = (_Float16)((float)E1[o] * (float)E2[o] + v);
      }
    }
  }
}

// ---------------------------------------------------------------------------
// Fused windowed attention: one block per (batch, window), 128 threads =
// 4 waves, wave h owns head h.
//  phase 0: all threads project LN(pair_cond) tile (f16, precomputed) through
//           this layer's Wb_pair into 4 per-head LDS bias planes
//  phase 1: per head: logits = Q K^T / sqrt(32) + bias   (WMMA, K=DH=32)
//  phase 2: in-place f16 softmax over 128 keys
//  phase 3: O = P V (WMMA, K=128), gate applied, store go16
// LDS rows padded to LDP_=144 halves (32B-aligned, bank-rotating).
// ---------------------------------------------------------------------------
__global__ __launch_bounds__(128) void attn_fused_kernel(
    const _Float16* __restrict__ q16,
    const _Float16* __restrict__ k16,
    const _Float16* __restrict__ v16,
    const _Float16* __restrict__ gate16,
    const _Float16* __restrict__ pcln,    // (B, NWIN, WQ, WK, 16) f16, LN'd
    const float*    __restrict__ Wbp,     // (16, 4) f32
    _Float16* __restrict__ go16)
{
  extern __shared__ char smem[];
  _Float16* slogBase = (_Float16*)smem;                       // H * WQ * LDP
  _Float16* svTBase  = slogBase + H_ * WQ_ * LDP_;            // H * DH * LDP

  const int tid  = threadIdx.x;
  const int hh   = tid >> 5;          // wave id == head
  const int lane = tid & 31;
  const int l16  = lane & 15;
  const int hi   = lane >> 4;
  const int w    = blockIdx.x % NWIN_;
  const int b    = blockIdx.x / NWIN_;
  const size_t tb = (size_t)b * NTOK_;

  _Float16* slogp = slogBase + hh * (WQ_ * LDP_);
  _Float16* svTp  = svTBase  + hh * (DH_ * LDP_);

  // Load the tiny Wb_pair into registers (uniform across the block)
  float wb[DP_ * H_];
  #pragma unroll
  for (int i = 0; i < DP_ * H_; i++) wb[i] = Wbp[i];

  // --- phase 0: pair bias -> all 4 head planes (cooperative, all threads) ---
  {
    const size_t rbase = (size_t)(b * NWIN_ + w) * (WQ_ * WK_);
    for (int i = tid; i < WQ_ * WK_; i += 128) {
      int qi = i >> 7, kj = i & 127;
      h16v pr = *(const h16v*)&pcln[(rbase + i) * DP_];
      float o0 = 0.f, o1 = 0.f, o2 = 0.f, o3 = 0.f;
      #pragma unroll
      for (int d = 0; d < DP_; d++) {
        float x = (float)pr[d];
        o0 += x * wb[d * 4 + 0];
        o1 += x * wb[d * 4 + 1];
        o2 += x * wb[d * 4 + 2];
        o3 += x * wb[d * 4 + 3];
      }
      int ofs = qi * LDP_ + kj;
      slogBase[0 * WQ_ * LDP_ + ofs] = (_Float16)o0;
      slogBase[1 * WQ_ * LDP_ + ofs] = (_Float16)o1;
      slogBase[2 * WQ_ * LDP_ + ofs] = (_Float16)o2;
      slogBase[3 * WQ_ * LDP_ + ofs] = (_Float16)o3;
    }
  }

  // --- stage V^T for this head: svT[c*LDP + kj] = V[pos(kj)][hh*32+c] ---
  for (int kj = lane; kj < WK_; kj += 32) {
    int pos = w * WQ_ - 32 + kj;
    if (pos >= 0 && pos < NTOK_) {
      const _Float16* src = v16 + (tb + pos) * D_ + hh * DH_;
      #pragma unroll
      for (int c = 0; c < DH_; c++) svTp[c * LDP_ + kj] = src[c];
    } else {
      #pragma unroll
      for (int c = 0; c < DH_; c++) svTp[c * LDP_ + kj] = (_Float16)0.f;
    }
  }
  __syncthreads();

  // --- phase 1: logits = QK^T * scale + bias (in place over bias plane) ---
  h16v qa[4];
  #pragma unroll
  for (int mt = 0; mt < 4; mt++)
    qa[mt] = *(const h16v*)(q16 + (tb + w * WQ_ + mt * 16 + l16) * D_ + hh * DH_ + hi * 16);

  const float scale = 0.17677669529663689f;   // 1/sqrt(32)
  for (int nt = 0; nt < 8; nt++) {
    int pos = w * WQ_ - 32 + nt * 16 + l16;
    h16v kb = {};
    if (pos >= 0 && pos < NTOK_)
      kb = *(const h16v*)(k16 + (tb + pos) * D_ + hh * DH_ + hi * 16);
    #pragma unroll
    for (int mt = 0; mt < 4; mt++) {
      f8v z = (f8v){0.f,0.f,0.f,0.f,0.f,0.f,0.f,0.f};
      f8v a = wmma16(qa[mt], kb, z);
      #pragma unroll
      for (int r = 0; r < 8; r++) {
        int qi  = mt * 16 + r + hi * 8;
        int kj  = nt * 16 + l16;
        int ofs = qi * LDP_ + kj;
        slogp[ofs] = (_Float16)(a[r] * scale + (float)slogp[ofs]);
      }
    }
  }
  __syncthreads();

  // --- phase 2: in-place row softmax (f32 math on f16 storage) ---
  for (int rr = lane; rr < WQ_; rr += 32) {
    float m = -1e30f;
    for (int j = 0; j < WK_; j++) m = fmaxf(m, (float)slogp[rr * LDP_ + j]);
    float s = 0.f;
    for (int j = 0; j < WK_; j++) s += __expf((float)slogp[rr * LDP_ + j] - m);
    float inv = 1.f / s;
    for (int j = 0; j < WK_; j++)
      slogp[rr * LDP_ + j] =
          (_Float16)(__expf((float)slogp[rr * LDP_ + j] - m) * inv);
  }
  __syncthreads();

  // --- phase 3: O = P V (K=128 -> 4 chained WMMAs), gate + store ---
  #pragma unroll
  for (int mt = 0; mt < 4; mt++) {
    #pragma unroll
    for (int ot = 0; ot < 2; ot++) {
      f8v acc = (f8v){0.f,0.f,0.f,0.f,0.f,0.f,0.f,0.f};
      #pragma unroll
      for (int kt = 0; kt < 4; kt++) {
        h16v af = *(const h16v*)&slogp[(mt * 16 + l16) * LDP_ + kt * 32 + hi * 16];
        h16v vf = *(const h16v*)&svTp [(ot * 16 + l16) * LDP_ + kt * 32 + hi * 16];
        acc = wmma16(af, vf, acc);
      }
      #pragma unroll
      for (int r = 0; r < 8; r++) {
        int t   = w * WQ_ + mt * 16 + r + hi * 8;
        int col = hh * DH_ + ot * 16 + l16;
        size_t o = (tb + t) * D_ + col;
        go16[o] = (_Float16)((float)gate16[o] * acc[r]);
      }
    }
  }
}

// ---------------------------------------------------------------------------
// Row LayerNorm -> f16.  One wave per row (4 f32 per lane), 8 rows per block.
// Rows with tl >= ninPerB produce zeros (matches LN of zero padding).
// ---------------------------------------------------------------------------
__global__ __launch_bounds__(256) void rowln_kernel(
    const float* __restrict__ in, _Float16* __restrict__ out, int ninPerB)
{
  int row  = blockIdx.x * 8 + (threadIdx.x >> 5);
  int lane = threadIdx.x & 31;
  int b = row / NTOK_, tl = row % NTOK_;
  float v0 = 0.f, v1 = 0.f, v2 = 0.f, v3 = 0.f;
  if (tl < ninPerB) {
    const float* s = in + ((size_t)b * ninPerB + tl) * D_ + lane * 4;
    v0 = s[0]; v1 = s[1]; v2 = s[2]; v3 = s[3];
  }
  float s = v0 + v1 + v2 + v3;
  float q = v0 * v0 + v1 * v1 + v2 * v2 + v3 * v3;
  #pragma unroll
  for (int m = 16; m > 0; m >>= 1) {
    s += __shfl_xor(s, m, 32);
    q += __shfl_xor(q, m, 32);
  }
  float mean = s * (1.f / D_);
  float var  = q * (1.f / D_) - mean * mean;
  float rs   = rsqrtf(var + 1e-5f);
  _Float16* d = out + (size_t)row * D_ + lane * 4;
  d[0] = (_Float16)((v0 - mean) * rs);
  d[1] = (_Float16)((v1 - mean) * rs);
  d[2] = (_Float16)((v2 - mean) * rs);
  d[3] = (_Float16)((v3 - mean) * rs);
}

// ---------------------------------------------------------------------------
// LN(pair_cond) -> f16, computed ONCE (layer-invariant).  Streams 266MB f32
// in, 133MB f16 out; one thread per 16-wide row with speculative prefetch.
// ---------------------------------------------------------------------------
__global__ __launch_bounds__(256) void pair_ln_kernel(
    const float* __restrict__ pc, _Float16* __restrict__ pcln)
{
  size_t r = (size_t)blockIdx.x * 256 + threadIdx.x;   // B*NWIN*WQ*WK rows
  const float* src = pc + r * DP_;
  __builtin_prefetch(src + DP_ * 64, 0, 1);            // global_prefetch_b8
  float v[DP_]; float s = 0.f;
  #pragma unroll
  for (int i = 0; i < DP_; i++) { v[i] = src[i]; s += v[i]; }
  float mean = s * (1.f / DP_);
  float q = 0.f;
  #pragma unroll
  for (int i = 0; i < DP_; i++) { float d = v[i] - mean; q += d * d; }
  float rs = rsqrtf(q * (1.f / DP_) + 1e-5f);
  h16v o;
  #pragma unroll
  for (int i = 0; i < DP_; i++) o[i] = (_Float16)((v[i] - mean) * rs);
  *(h16v*)&pcln[r * DP_] = o;
}

// --------------------------- small utility kernels -------------------------
__global__ __launch_bounds__(256) void wconvT_kernel(
    const float* __restrict__ src, _Float16* __restrict__ dst, int K, int Ncols)
{
  int i = blockIdx.x * 256 + threadIdx.x;
  if (i >= K * Ncols) return;
  int k = i / Ncols, n = i % Ncols;
  dst[(size_t)n * K + k] = (_Float16)src[i];   // (K,N) f32 -> (N,K) f16
}

__global__ __launch_bounds__(256) void pad_copy_kernel(
    const float* __restrict__ src, float* __restrict__ dst)
{
  size_t i = (size_t)blockIdx.x * 256 + threadIdx.x;   // B*NTOK*D
  size_t c = i % D_; size_t t = i / D_;
  size_t tl = t % NTOK_; size_t b = t / NTOK_;
  dst[i] = (tl < N_) ? src[((size_t)b * N_ + tl) * D_ + c] : 0.f;
}

__global__ __launch_bounds__(256) void combine_kernel(
    float* __restrict__ x, const float* __restrict__ oproj,
    const float* __restrict__ ffo,
    const _Float16* __restrict__ ga, const _Float16* __restrict__ gf)
{
  size_t i = (size_t)blockIdx.x * 256 + threadIdx.x;   // MT_*D
  x[i] += (float)ga[i] * oproj[i] + (float)gf[i] * ffo[i];
}

__global__ __launch_bounds__(256) void final_copy_kernel(
    const float* __restrict__ x, float* __restrict__ out)
{
  size_t i = (size_t)blockIdx.x * 256 + threadIdx.x;
  if (i >= (size_t)B_ * N_ * D_) return;
  size_t rem = i % ((size_t)N_ * D_);
  size_t b   = i / ((size_t)N_ * D_);
  out[i] = x[b * (size_t)NTOK_ * D_ + rem];
}

// ---------------------------------------------------------------------------
extern "C" void kernel_launch(void* const* d_in, const int* in_sizes, int n_in,
                              void* d_out, int out_size, void* d_ws, size_t ws_size,
                              hipStream_t stream)
{
  (void)in_sizes; (void)n_in; (void)out_size; (void)ws_size;

  const float* single_repr = (const float*)d_in[0];
  const float* single_cond = (const float*)d_in[1];
  const float* pair_cond   = (const float*)d_in[2];

  // Per-layer param leaves in jax dict-sorted key order:
  enum { iW1 = 0, iW2, iWb_pair, iWcg, iWgate, iWk, iWo, iWout, iWq, iWv,
         ia_Wb, ia_Wg, ia_bg, ibcg, ibq, if_Wb, if_Wcg, if_Wg, if_bcg, if_bg };
  auto PL = [&](int layer, int idx) -> const float* {
    return (const float*)d_in[3 + layer * 20 + idx];
  };

  // Workspace bump allocator
  char* base = (char*)d_ws;
  size_t cur = 0;
  auto alloc = [&](size_t bytes) -> void* {
    cur = (cur + 255) & ~(size_t)255;
    void* p = base + cur; cur += bytes; return p;
  };
  const size_t SZ16   = (size_t)MT_ * D_ * sizeof(_Float16);
  const size_t SZ32   = (size_t)MT_ * D_ * sizeof(float);
  const size_t SZFF16 = (size_t)MT_ * FF_ * sizeof(_Float16);
  const size_t NPAIR  = (size_t)B_ * NWIN_ * WQ_ * WK_;     // 4,161,536 rows

  float*     x       = (float*)alloc(SZ32);
  _Float16*  condln  = (_Float16*)alloc(SZ16);
  _Float16*  xln     = (_Float16*)alloc(SZ16);
  _Float16*  tA      = (_Float16*)alloc(SZ16);
  _Float16*  a16     = (_Float16*)alloc(SZ16);
  _Float16*  af16    = (_Float16*)alloc(SZ16);
  _Float16*  q16     = (_Float16*)alloc(SZ16);
  _Float16*  k16     = (_Float16*)alloc(SZ16);
  _Float16*  v16b    = (_Float16*)alloc(SZ16);
  _Float16*  gate16  = (_Float16*)alloc(SZ16);
  _Float16*  ga16    = (_Float16*)alloc(SZ16);
  _Float16*  gf16    = (_Float16*)alloc(SZ16);
  _Float16*  s1h     = (_Float16*)alloc(SZFF16);
  _Float16*  go16    = (_Float16*)alloc(SZ16);
  float*     oproj   = (float*)alloc(SZ32);
  float*     ffo     = (float*)alloc(SZ32);
  _Float16*  pcln16  = (_Float16*)alloc(NPAIR * DP_ * sizeof(_Float16));  // 133MB
  // transposed f16 weights (reused across layers)
  _Float16* wt_aWg  = (_Float16*)alloc(128 * 128 * 2);
  _Float16* wt_aWb  = (_Float16*)alloc(128 * 128 * 2);
  _Float16* wt_Wq   = (_Float16*)alloc(128 * 128 * 2);
  _Float16* wt_Wk   = (_Float16*)alloc(128 * 128 * 2);
  _Float16* wt_Wv   = (_Float16*)alloc(128 * 128 * 2);
  _Float16* wt_Wg8  = (_Float16*)alloc(128 * 128 * 2);   // Wgate
  _Float16* wt_Wo   = (_Float16*)alloc(128 * 128 * 2);
  _Float16* wt_Wcg  = (_Float16*)alloc(128 * 128 * 2);
  _Float16* wt_fWg  = (_Float16*)alloc(128 * 128 * 2);
  _Float16* wt_fWb  = (_Float16*)alloc(128 * 128 * 2);
  _Float16* wt_fWcg = (_Float16*)alloc(128 * 128 * 2);
  _Float16* wt_W1   = (_Float16*)alloc(256 * 128 * 2);   // (FF,K) after transpose
  _Float16* wt_W2   = (_Float16*)alloc(256 * 128 * 2);
  _Float16* wt_Wout = (_Float16*)alloc(128 * 256 * 2);   // (D,FF)

  const int GGRID = MT_ / 128;                 // 254 blocks per GEMM
  const int EGRID = (int)(((size_t)MT_ * D_) / 256);
  const int ATTN_LDS = (H_ * WQ_ * LDP_ + H_ * DH_ * LDP_) * (int)sizeof(_Float16);

  auto wconv = [&](const float* src, _Float16* dst, int K, int Ncols) {
    int total = K * Ncols;
    wconvT_kernel<<<(total + 255) / 256, 256, 0, stream>>>(src, dst, K, Ncols);
  };

  // ---- init: pad x, LN(single_cond), LN(pair_cond) (once, layer-invariant) ----
  pad_copy_kernel<<<EGRID, 256, 0, stream>>>(single_repr, x);
  rowln_kernel<<<MT_ / 8, 256, 0, stream>>>(single_cond, condln, N_);
  pair_ln_kernel<<<(int)(NPAIR / 256), 256, 0, stream>>>(pair_cond, pcln16);

  for (int l = 0; l < DEPTH_; l++) {
    // weight prep (f32 -> f16 transposed)
    wconv(PL(l, ia_Wg),  wt_aWg,  128, 128);
    wconv(PL(l, ia_Wb),  wt_aWb,  128, 128);
    wconv(PL(l, iWq),    wt_Wq,   128, 128);
    wconv(PL(l, iWk),    wt_Wk,   128, 128);
    wconv(PL(l, iWv),    wt_Wv,   128, 128);
    wconv(PL(l, iWgate), wt_Wg8,  128, 128);
    wconv(PL(l, iWo),    wt_Wo,   128, 128);
    wconv(PL(l, iWcg),   wt_Wcg,  128, 128);
    wconv(PL(l, if_Wg),  wt_fWg,  128, 128);
    wconv(PL(l, if_Wb),  wt_fWb,  128, 128);
    wconv(PL(l, if_Wcg), wt_fWcg, 128, 128);
    wconv(PL(l, iW1),    wt_W1,   128, 256);
    wconv(PL(l, iW2),    wt_W2,   128, 256);
    wconv(PL(l, iWout),  wt_Wout, 256, 128);

    // LN(x) -> xln
    rowln_kernel<<<MT_ / 8, 256, 0, stream>>>(x, xln, NTOK_);

    // adaLN a: a = sigmoid(condln@aWg + bg) * xln + condln@aWb
    gemm_wmma_k<128, EP_SIGBIAS><<<GGRID, 256, 0, stream>>>(
        condln, wt_aWg, PL(l, ia_bg), nullptr, nullptr, tA, 128);
    gemm_wmma_k<128, EP_MULADD><<<GGRID, 256, 0, stream>>>(
        condln, wt_aWb, nullptr, tA, xln, a16, 128);
    // adaLN f
    gemm_wmma_k<128, EP_SIGBIAS><<<GGRID, 256, 0, stream>>>(
        condln, wt_fWg, PL(l, if_bg), nullptr, nullptr, tA, 128);
    gemm_wmma_k<128, EP_MULADD><<<GGRID, 256, 0, stream>>>(
        condln, wt_fWb, nullptr, tA, xln, af16, 128);

    // projections
    gemm_wmma_k<128, EP_BIAS><<<GGRID, 256, 0, stream>>>(
        a16, wt_Wq, PL(l, ibq), nullptr, nullptr, q16, 128);
    gemm_wmma_k<128, EP_F16><<<GGRID, 256, 0, stream>>>(
        a16, wt_Wk, nullptr, nullptr, nullptr, k16, 128);
    gemm_wmma_k<128, EP_F16><<<GGRID, 256, 0, stream>>>(
        a16, wt_Wv, nullptr, nullptr, nullptr, v16b, 128);
    gemm_wmma_k<128, EP_SIG><<<GGRID, 256, 0, stream>>>(
        a16, wt_Wg8, nullptr, nullptr, nullptr, gate16, 128);
    // channel gates
    gemm_wmma_k<128, EP_SIGBIAS><<<GGRID, 256, 0, stream>>>(
        condln, wt_Wcg, PL(l, ibcg), nullptr, nullptr, ga16, 128);
    gemm_wmma_k<128, EP_SIGBIAS><<<GGRID, 256, 0, stream>>>(
        condln, wt_fWcg, PL(l, if_bcg), nullptr, nullptr, gf16, 128);

    // fused attention (pair bias projection + QK^T + softmax + PV + gate)
    attn_fused_kernel<<<B_ * NWIN_, 128, ATTN_LDS, stream>>>(
        q16, k16, v16b, gate16, pcln16, PL(l, iWb_pair), go16);

    // output projection
    gemm_wmma_k<128, EP_F32><<<GGRID, 256, 0, stream>>>(
        go16, wt_Wo, nullptr, nullptr, nullptr, oproj, 128);

    // FF: h = silu(af@W1) * (af@W2), in column halves of 128
    for (int ch = 0; ch < 2; ch++) {
      gemm_wmma_k<128, EP_SILU><<<GGRID, 256, 0, stream>>>(
          af16, wt_W1 + (size_t)ch * 128 * 128, nullptr, nullptr, nullptr,
          s1h + ch * 128, FF_);
    }
    for (int ch = 0; ch < 2; ch++) {
      gemm_wmma_k<128, EP_MUL><<<GGRID, 256, 0, stream>>>(
          af16, wt_W2 + (size_t)ch * 128 * 128, nullptr, s1h + ch * 128, nullptr,
          s1h + ch * 128, FF_);
    }
    gemm_wmma_k<256, EP_F32><<<GGRID, 256, 0, stream>>>(
        s1h, wt_Wout, nullptr, nullptr, nullptr, ffo, 128);

    // x += ga*oproj + gf*ffo
    combine_kernel<<<EGRID, 256, 0, stream>>>(x, oproj, ffo, ga16, gf16);
  }

  // ---- output: x[:, :N, :] ----
  size_t outElems = (size_t)B_ * N_ * D_;
  final_copy_kernel<<<(int)((outElems + 255) / 256), 256, 0, stream>>>(
      x, (float*)d_out);
}
